// GatedSwitchGNN_73031623901502
// MI455X (gfx1250) — matrix operand order repaired and checked by reading, not compiled, not run on hardware
//
#include <hip/hip_runtime.h>

#define Bsz 200
#define Nn 48
#define Ff 96
#define NSWc 8
#define NEc 47
#define Mc 55
#define ESTR 98   // padded LDS row stride (floats) to avoid bank conflicts

typedef unsigned short ushort_t;
typedef __attribute__((ext_vector_type(16))) __bf16 v16bf;
typedef __attribute__((ext_vector_type(8)))  float  v8f;
typedef __attribute__((ext_vector_type(2)))  float  v2f;

union Frag16 { v16bf v; uint4 q[2]; };

__device__ __forceinline__ ushort_t f2bf(float x) {
  unsigned u = __float_as_uint(x);
  u += 0x7FFFu + ((u >> 16) & 1u);
  return (ushort_t)(u >> 16);
}
__device__ __forceinline__ unsigned pack2bf(float a, float b) {
  return (unsigned)f2bf(a) | ((unsigned)f2bf(b) << 16);
}
__device__ __forceinline__ float bf2f(ushort_t h) {
  return __uint_as_float(((unsigned)h) << 16);
}
__device__ __forceinline__ float sigm(float x) { return 1.f / (1.f + __expf(-x)); }

// A-fragment: 16x32 bf16 tile from row-major bf16 activations, row stride ldA (halves).
// Contiguous k order used here (pack_b/pack_proj match): lanes 0-15 hold K 0..15, lanes 16-31 hold K 16..31.
__device__ __forceinline__ v16bf load_a_frag(const ushort_t* A, size_t row0, int ldA, int kc) {
  int lane = (int)(threadIdx.x & 31u);
  int r    = lane & 15;
  int half = lane >> 4;
  const ushort_t* p = A + (row0 + (size_t)r) * (size_t)ldA + kc * 32 + half * 16;
  Frag16 f;
  f.q[0] = *(const uint4*)(p);
  f.q[1] = *(const uint4*)(p + 8);
  return f.v;
}
// B-fragment from pre-packed weights: 16 contiguous halves per lane.
__device__ __forceinline__ v16bf load_b_frag(const ushort_t* Bp, int nt, int kc, int KC) {
  int lane = (int)(threadIdx.x & 31u);
  const ushort_t* p = Bp + ((size_t)(nt * KC + kc) * 32 + (size_t)lane) * 16;
  Frag16 f;
  f.q[0] = *(const uint4*)(p);
  f.q[1] = *(const uint4*)(p + 8);
  return f.v;
}
__device__ __forceinline__ v8f wmma_bf16(v16bf a, v16bf b, v8f c) {
  return __builtin_amdgcn_wmma_f32_16x16x32_bf16(false, a, false, b, (short)0, c, false, false);
}

// ---------------- setup kernels ----------------

__global__ void init_lookup(const int* sw_i, const int* sw_j,
                            int* pairSlot, int* nodePair, int* nodePartner) {
  int t = (int)threadIdx.x;
  for (int idx = t; idx < Nn * Nn; idx += (int)blockDim.x) pairSlot[idx] = -1;
  for (int idx = t; idx < Nn; idx += (int)blockDim.x) { nodePair[idx] = -1; nodePartner[idx] = 0; }
  __syncthreads();
  if (t < NSWc) {
    int i = sw_i[t], j = sw_j[t];
    pairSlot[i * Nn + j] = t;
    pairSlot[j * Nn + i] = t + NSWc;
    nodePair[i] = t;        nodePartner[i] = j;
    nodePair[j] = t + NSWc; nodePartner[j] = i;
  }
}

// s0[b,i,j,h] = emb[int(S[b,i,j])][h], bf16, K padded to 32; packed b32 stores
__global__ void build_s0(const float* S, const float* emb, ushort_t* s0) {
  size_t total = (size_t)Bsz * Nn * Nn * 16;   // half-pairs
  size_t stride = (size_t)gridDim.x * blockDim.x;
  for (size_t idx = (size_t)blockIdx.x * blockDim.x + threadIdx.x; idx < total; idx += stride) {
    size_t row = idx >> 4;
    int hp = (int)(idx & 15) * 2;
    unsigned pk = 0u;
    if (hp < 16) {
      int si = (int)S[row];
      pk = pack2bf(emb[si * 16 + hp], emb[si * 16 + hp + 1]);
    }
    *(unsigned*)(s0 + row * 32 + hp) = pk;
  }
}

// x (B,48,16) f32 -> bf16 padded to K=32; packed stores
__global__ void convert_x0(const float* x, ushort_t* xb) {
  size_t total = (size_t)Bsz * Nn * 16;
  size_t stride = (size_t)gridDim.x * blockDim.x;
  for (size_t idx = (size_t)blockIdx.x * blockDim.x + threadIdx.x; idx < total; idx += stride) {
    size_t row = idx >> 4;
    int hp = (int)(idx & 15) * 2;
    unsigned pk = 0u;
    if (hp < 16) pk = pack2bf(x[row * 16 + hp], x[row * 16 + hp + 1]);
    *(unsigned*)(xb + row * 32 + hp) = pk;
  }
}

// Pack a (finRows x Ncols) f32 weight into WMMA-B bf16 layout, K padded to Kpad.
__global__ void pack_b(const float* W, int finRows, int Kpad, int Ncols, ushort_t* out) {
  int total = Kpad * Ncols;
  int KC = Kpad / 32;
  int stride = (int)(gridDim.x * blockDim.x);
  for (int idx = (int)(blockIdx.x * blockDim.x + threadIdx.x); idx < total; idx += stride) {
    int K = idx / Ncols, c = idx % Ncols;
    float v = (K < finRows) ? W[K * Ncols + c] : 0.f;
    int kc = K / 32, kk = K % 32;
    int lane = ((kk < 16) ? 0 : 16) + (c & 15);
    int id16 = kk & 15;
    int nt = c >> 4;
    out[(((size_t)nt * KC + kc) * 32 + lane) * 16 + id16] = f2bf(v);
  }
}

// Pack [Wu|Wv|Wa|Wb] (each finRows x 96) into one 384-col WMMA-B bf16 matrix.
__global__ void pack_proj(const float* Wu, const float* Wv, const float* Wa, const float* Wb,
                          int finRows, int Kpad, ushort_t* out) {
  int total = Kpad * 384;
  int KC = Kpad / 32;
  int stride = (int)(gridDim.x * blockDim.x);
  for (int idx = (int)(blockIdx.x * blockDim.x + threadIdx.x); idx < total; idx += stride) {
    int K = idx / 384, c = idx % 384;
    const float* W = (c < 96) ? Wu : (c < 192) ? Wv : (c < 288) ? Wa : Wb;
    int cc = c % 96;
    float v = (K < finRows) ? W[K * 96 + cc] : 0.f;
    int kc = K / 32, kk = K % 32;
    int lane = ((kk < 16) ? 0 : 16) + (c & 15);
    int id16 = kk & 15;
    int nt = c >> 4;
    out[(((size_t)nt * KC + kc) * 32 + lane) * 16 + id16] = f2bf(v);
  }
}

// ---------------- generic bf16 WMMA GEMM (rows x Kpad) @ packed(Kpad x ntiles*16) ----------------
template <int NTPW>
__global__ void gemm_rows(const ushort_t* Abf, int ldA, const ushort_t* Bp, int KC,
                          const float* bias, int doRelu, float* C, int ldC) {
  int wave = (int)(threadIdx.x >> 5);
  int lane = (int)(threadIdx.x & 31u);
  size_t row0 = (size_t)blockIdx.x * 16;
  v8f acc[NTPW] = {};
  for (int kc = 0; kc < KC; ++kc) {
    v16bf a = load_a_frag(Abf, row0, ldA, kc);
#pragma unroll
    for (int t = 0; t < NTPW; ++t) {
      v16bf b = load_b_frag(Bp, wave * NTPW + t, kc, KC);
      acc[t] = wmma_bf16(a, b, acc[t]);
    }
  }
  int col = lane & 15;
  int rbase = (lane < 16) ? 0 : 8;
#pragma unroll
  for (int t = 0; t < NTPW; ++t) {
    int f = (wave * NTPW + t) * 16 + col;
    float bv = bias ? bias[f] : 0.f;
#pragma unroll
    for (int r = 0; r < 8; ++r) {
      float v = acc[t][r] + bv;
      if (doRelu) v = v > 0.f ? v : 0.f;
      C[(row0 + (size_t)(rbase + r)) * (size_t)ldC + f] = v;
    }
  }
}

// ---------------- edge kernel: e = s@We + xa_i + xb_j; sn = [s+] relu(LN(e)); gate at switch pairs ----
__global__ void edge_kernel(const ushort_t* sIn, int ldS, int KC,
                            const ushort_t* WeP, const float* proj,
                            ushort_t* sOut, float* gate, const int* pairSlot, int residual) {
  __shared__ float eT[16 * ESTR];
  __shared__ float mRow[16], rRow[16];
  __shared__ int   slotRow[16];
  int rt = (int)blockIdx.x;               // 0..28799
  int b = rt / 144;
  int rem = rt - b * 144;
  int i = rem / 3;
  int j0 = (rem % 3) * 16;
  int wave = (int)(threadIdx.x >> 5);
  int lane = (int)(threadIdx.x & 31u);
  size_t row0 = (size_t)rt * 16;

  // Prefetch the residual tile (read again in the tail) so it overlaps the WMMA loop.
  if (residual) {
    for (int t = (int)threadIdx.x; t < 32; t += (int)blockDim.x) {
      int row = t >> 1, off = (t & 1) * 64;
      __builtin_prefetch(sIn + (row0 + (size_t)row) * (size_t)ldS + off, 0, 3);
    }
  }

  v8f acc[3] = {};
  for (int kc = 0; kc < KC; ++kc) {
    v16bf a = load_a_frag(sIn, row0, ldS, kc);
#pragma unroll
    for (int t = 0; t < 3; ++t) {
      v16bf bb = load_b_frag(WeP, wave * 3 + t, kc, KC);
      acc[t] = wmma_bf16(a, bb, acc[t]);
    }
  }
  int col = lane & 15;
  int rbase = (lane < 16) ? 0 : 8;
  const float* xa  = proj + ((size_t)(b * Nn + i)) * 384 + 192;   // x@Wa row i
  const float* xbp = proj + ((size_t)(b * Nn)) * 384 + 288;       // x@Wb rows (stride 384)
#pragma unroll
  for (int t = 0; t < 3; ++t) {
    int f = (wave * 3 + t) * 16 + col;
    float xav = xa[f];
#pragma unroll
    for (int r = 0; r < 8; ++r) {
      int row = rbase + r;
      int j = j0 + row;
      eT[row * ESTR + f] = acc[t][r] + xav + xbp[(size_t)j * 384 + f];
    }
  }
  __syncthreads();
  if (threadIdx.x < 16) {
    int row = (int)threadIdx.x;
    float m = 0.f;
    for (int f = 0; f < 96; ++f) m += eT[row * ESTR + f];
    m *= (1.f / 96.f);
    float var = 0.f;
    for (int f = 0; f < 96; ++f) { float d = eT[row * ESTR + f] - m; var += d * d; }
    mRow[row] = m;
    rRow[row] = rsqrtf(var * (1.f / 96.f) + 1e-5f);
    slotRow[row] = pairSlot[i * Nn + (j0 + row)];
  }
  __syncthreads();
  // packed (2 elems / thread) write-out: b32 bf16 stores, b64 gate stores
  for (int idx = (int)threadIdx.x; idx < 16 * 48; idx += (int)blockDim.x) {
    int row = idx / 48;
    int fp = (idx - row * 48) * 2;
    size_t gr = row0 + (size_t)row;
    float e0 = eT[row * ESTR + fp];
    float e1 = eT[row * ESTR + fp + 1];
    float sn0 = (e0 - mRow[row]) * rRow[row]; sn0 = sn0 > 0.f ? sn0 : 0.f;
    float sn1 = (e1 - mRow[row]) * rRow[row]; sn1 = sn1 > 0.f ? sn1 : 0.f;
    if (residual) {
      unsigned pr = *(const unsigned*)(sIn + gr * (size_t)ldS + fp);
      sn0 += bf2f((ushort_t)(pr & 0xFFFFu));
      sn1 += bf2f((ushort_t)(pr >> 16));
    }
    *(unsigned*)(sOut + gr * 96 + fp) = pack2bf(sn0, sn1);
    int pp = slotRow[row];
    if (pp >= 0) {
      float2 g; g.x = sigm(e0); g.y = sigm(e1);
      *(float2*)(gate + ((size_t)b * 16 + pp) * 96 + fp) = g;
    }
  }
}

// ---------------- node kernel: xn = [x+] relu(LN(Ux + A@Vx + S.gate.Vx)) ----------------
// fp32 WMMA 16x16x4 for A@Vx; one workgroup (4 waves) per batch
__global__ void node_kernel(const float* Aall, const float* proj, const float* gate,
                            const int* nodePair, const int* nodePartner,
                            const float* xPrev, int residual,
                            float* xOutF, ushort_t* xOutB) {
  __shared__ float tX[Nn * ESTR];
  __shared__ float mRow[Nn], rRow[Nn];
  int b = (int)blockIdx.x;
  int wave = (int)(threadIdx.x >> 5);
  int lane = (int)(threadIdx.x & 31u);
  const float* Ab = Aall + (size_t)b * Nn * Nn;
  const float* Vx = proj + (size_t)b * Nn * 384 + 96;   // row stride 384

  int r15 = lane & 15;
  int khalf = (lane < 16) ? 0 : 2;
  for (int pr = wave; pr < 18; pr += 4) {
    int mt = pr / 6, nt = pr - mt * 6;
    v8f acc = {};
    for (int kc = 0; kc < 12; ++kc) {
      int k0 = kc * 4 + khalf;
      v2f a;  a.x = Ab[(mt * 16 + r15) * Nn + k0];         a.y = Ab[(mt * 16 + r15) * Nn + k0 + 1];
      v2f bb; bb.x = Vx[(size_t)k0 * 384 + nt * 16 + r15]; bb.y = Vx[(size_t)(k0 + 1) * 384 + nt * 16 + r15];
      acc = __builtin_amdgcn_wmma_f32_16x16x4_f32(false, a, false, bb, (short)0, acc, false, false);
    }
    int rbase = (lane < 16) ? 0 : 8;
#pragma unroll
    for (int r = 0; r < 8; ++r) {
      tX[(mt * 16 + rbase + r) * ESTR + nt * 16 + r15] = acc[r];
    }
  }
  __syncthreads();
  if (threadIdx.x < Nn) {
    int i = (int)threadIdx.x;
    const float* Uxr = proj + ((size_t)(b * Nn + i)) * 384;     // Ux at cols 0..95
    int pp = nodePair[i];
    int jj = nodePartner[i];
    float m = 0.f;
    for (int f = 0; f < 96; ++f) {
      float v = tX[i * ESTR + f] + Uxr[f];
      if (pp >= 0) v += gate[((size_t)b * 16 + pp) * 96 + f] * Vx[(size_t)jj * 384 + f];
      tX[i * ESTR + f] = v;
      m += v;
    }
    m *= (1.f / 96.f);
    float var = 0.f;
    for (int f = 0; f < 96; ++f) { float d = tX[i * ESTR + f] - m; var += d * d; }
    mRow[i] = m;
    rRow[i] = rsqrtf(var * (1.f / 96.f) + 1e-5f);
  }
  __syncthreads();
  for (int idx = (int)threadIdx.x; idx < Nn * 48; idx += (int)blockDim.x) {
    int i = idx / 48;
    int fp = (idx - i * 48) * 2;
    size_t base = ((size_t)b * Nn + i) * 96 + fp;
    float v0 = (tX[i * ESTR + fp]     - mRow[i]) * rRow[i]; v0 = v0 > 0.f ? v0 : 0.f;
    float v1 = (tX[i * ESTR + fp + 1] - mRow[i]) * rRow[i]; v1 = v1 > 0.f ? v1 : 0.f;
    if (residual) {
      float2 xp = *(const float2*)(xPrev + base);
      v0 += xp.x; v1 += xp.y;
    }
    float2 o; o.x = v0; o.y = v1;
    *(float2*)(xOutF + base) = o;
    *(unsigned*)(xOutB + base) = pack2bf(v0, v1);
  }
}

// ---------------- heads ----------------

__global__ void xg_kernel(const float* x, float* xg) {
  int b = (int)blockIdx.x;
  int f = (int)threadIdx.x;   // 96 threads
  float s = 0.f;
  for (int i = 0; i < Nn; ++i) s += x[((size_t)b * Nn + i) * 96 + f];
  xg[b * 96 + f] = s;
}

__global__ void build_smlp_in(const ushort_t* s3, const float* x3, const float* xg,
                              const int* sw_i, const int* sw_j, ushort_t* out) {
  int total = Bsz * NSWc * 384;
  int stride = (int)(gridDim.x * blockDim.x);
  for (int idx = (int)(blockIdx.x * blockDim.x + threadIdx.x); idx < total; idx += stride) {
    int q = idx / 384, c = idx - q * 384;
    int b = q / NSWc, t = q - b * NSWc;
    float v;
    if (c < 96) {
      int i = sw_i[t], j = sw_j[t];
      v = bf2f(s3[(((size_t)b * (Nn * Nn)) + (size_t)i * Nn + j) * 96 + c]);
    } else if (c < 192) {
      v = x3[((size_t)b * Nn + sw_i[t]) * 96 + (c - 96)];
    } else if (c < 288) {
      v = x3[((size_t)b * Nn + sw_j[t]) * 96 + (c - 192)];
    } else {
      v = xg[b * 96 + (c - 288)];
    }
    out[idx] = f2bf(v);
  }
}

__global__ void build_cmlp_in(const float* x3, const float* xg,
                              const int* e_i, const int* e_j, ushort_t* out) {
  size_t total = (size_t)9408 * 288;
  size_t stride = (size_t)gridDim.x * blockDim.x;
  for (size_t idx = (size_t)blockIdx.x * blockDim.x + threadIdx.x; idx < total; idx += stride) {
    int q = (int)(idx / 288), c = (int)(idx - (size_t)q * 288);
    float v = 0.f;
    if (q < Bsz * NEc) {
      int b = q / NEc, t = q - b * NEc;
      if (c < 96)       v = x3[((size_t)b * Nn + e_i[t]) * 96 + c];
      else if (c < 192) v = x3[((size_t)b * Nn + e_j[t]) * 96 + (c - 96)];
      else              v = xg[b * 96 + (c - 192)];
    }
    out[idx] = f2bf(v);
  }
}

__global__ void mlp2_kernel(const float* hid, int K, int NO, int rows,
                            const float* W2, const float* b2, float* out) {
  int total = rows * NO;
  int stride = (int)(gridDim.x * blockDim.x);
  for (int idx = (int)(blockIdx.x * blockDim.x + threadIdx.x); idx < total; idx += stride) {
    int q = idx / NO, c = idx - q * NO;
    float o = b2[c];
    for (int k = 0; k < K; ++k) o += hid[(size_t)q * K + k] * W2[k * NO + c];
    out[idx] = o;
  }
}

// ---------------- final incidence-matrix algebra ----------------
__global__ void final_kernel(const float* smlp_out, const float* cmlp_out,
                             const float* Dinv, const float* Incp, const float* Incc,
                             float* z, float* zc) {
  __shared__ float topo[Mc], pf[Mc], vpar[Mc], vch[Mc], pfc[Mc], qfc[Mc];
  __shared__ float w48[Nn], vv[Nn];
  int b = (int)blockIdx.x;
  int t = (int)threadIdx.x;
  if (t < Mc) {
    float gt, psf = 0.f, vsp = 0.f, vsc = 0.f, pcf = 0.f, vcp = 0.f, vcc = 0.f;
    if (t < NEc) {
      const float* c = cmlp_out + ((size_t)b * NEc + t) * 3;
      gt = 0.f; pcf = c[0]; vcp = c[1]; vcc = c[2];
    } else {
      const float* s = smlp_out + ((size_t)b * NSWc + (t - NEc)) * 4;
      gt = sigm(s[0]); psf = s[1]; vsp = s[2]; vsc = s[3];
    }
    topo[t] = gt; pf[t] = psf + pcf; vpar[t] = vcp + vsp; vch[t] = vcc + vsc;
  }
  __syncthreads();
  if (t < Nn) {
    float a = 0.f;
    for (int m = 0; m < Mc; ++m) a += Incp[t * Mc + m] * vpar[m] + Incc[t * Mc + m] * vch[m];
    w48[t] = a;
  }
  __syncthreads();
  if (t < Nn) {
    float a = 0.f;
    for (int k = 0; k < Nn; ++k) a += Dinv[t * Nn + k] * w48[k];
    vv[t] = (t == 0) ? 1.f : a;
  }
  __syncthreads();
  if (t < Mc) {
    float a = 0.f;
    for (int n = 0; n < Nn; ++n) a += (Incp[n * Mc + t] - Incc[n * Mc + t]) * vv[n];
    pfc[t] = pf[t] * topo[t];
    qfc[t] = topo[t] * a;
  }
  __syncthreads();
  if (t < Nn) {
    float pg = 0.f, qg = 0.f;
    for (int m = 0; m < Mc; ++m) {
      float d = Incp[t * Mc + m] - Incc[t * Mc + m];
      pg += d * pfc[m];
      qg += d * qfc[m];
    }
    z[(size_t)b * 158 + 55 + t]  = vv[t];
    zc[(size_t)b * 151 + 55 + t] = pg;
    zc[(size_t)b * 151 + 103 + t] = qg;
  }
  if (t < Mc) {
    z[(size_t)b * 158 + t]       = pfc[t];
    z[(size_t)b * 158 + 103 + t] = topo[t];
    zc[(size_t)b * 151 + t]      = qfc[t];
  }
}

// ---------------- host orchestration ----------------
extern "C" void kernel_launch(void* const* d_in, const int* in_sizes, int n_in,
                              void* d_out, int out_size, void* d_ws, size_t ws_size,
                              hipStream_t stream) {
  (void)in_sizes; (void)n_in; (void)out_size; (void)ws_size;
  const float* x_in = (const float*)d_in[0];
  const float* A_in = (const float*)d_in[1];
  const float* S_in = (const float*)d_in[2];
  const int* sw_i = (const int*)d_in[3];
  const int* sw_j = (const int*)d_in[4];
  const int* e_i  = (const int*)d_in[5];
  const int* e_j  = (const int*)d_in[6];
  const float* emb = (const float*)d_in[7];
  const float* Wl[3][5];
  for (int l = 0; l < 3; ++l)
    for (int k = 0; k < 5; ++k)
      Wl[l][k] = (const float*)d_in[8 + 5 * l + k];   // Wu Wv Wa Wb We
  const float* sW1 = (const float*)d_in[23];
  const float* sb1 = (const float*)d_in[24];
  const float* sW2 = (const float*)d_in[25];
  const float* sb2 = (const float*)d_in[26];
  const float* cW1 = (const float*)d_in[27];
  const float* cb1 = (const float*)d_in[28];
  const float* cW2 = (const float*)d_in[29];
  const float* cb2 = (const float*)d_in[30];
  const float* Dinv = (const float*)d_in[31];
  const float* Incp = (const float*)d_in[32];
  const float* Incc = (const float*)d_in[33];

  char* wp = (char*)d_ws;
  auto alloc = [&](size_t bytes) -> void* {
    void* r = (void*)wp;
    wp += (bytes + 255) & ~((size_t)255);
    return r;
  };
  ushort_t* sBufA = (ushort_t*)alloc((size_t)Bsz * 2304 * 96 * 2);
  ushort_t* sBufB = (ushort_t*)alloc((size_t)Bsz * 2304 * 96 * 2);
  float*    xfA   = (float*)alloc((size_t)Bsz * Nn * 96 * 4);
  float*    xfB   = (float*)alloc((size_t)Bsz * Nn * 96 * 4);
  ushort_t* xbf   = (ushort_t*)alloc((size_t)Bsz * Nn * 96 * 2);
  float*    proj  = (float*)alloc((size_t)Bsz * Nn * 384 * 4);
  float*    gate  = (float*)alloc((size_t)Bsz * 16 * 96 * 4);
  ushort_t* WprojP = (ushort_t*)alloc((size_t)96 * 384 * 2);
  ushort_t* WeP    = (ushort_t*)alloc((size_t)96 * 96 * 2);
  ushort_t* W1sP   = (ushort_t*)alloc((size_t)384 * 384 * 2);
  ushort_t* W1cP   = (ushort_t*)alloc((size_t)288 * 288 * 2);
  ushort_t* smlpIn = (ushort_t*)alloc((size_t)1600 * 384 * 2);
  float*    smlpHid = (float*)alloc((size_t)1600 * 384 * 4);
  float*    smlpOut = (float*)alloc((size_t)1600 * 4 * 4);
  ushort_t* cmlpIn = (ushort_t*)alloc((size_t)9408 * 288 * 2);
  float*    cmlpHid = (float*)alloc((size_t)9408 * 288 * 4);
  float*    cmlpOut = (float*)alloc((size_t)Bsz * NEc * 3 * 4);
  float*    xg      = (float*)alloc((size_t)Bsz * 96 * 4);
  int*      pairSlot = (int*)alloc((size_t)Nn * Nn * 4);
  int*      nodePair = (int*)alloc((size_t)Nn * 4);
  int*      nodePartner = (int*)alloc((size_t)Nn * 4);

  init_lookup<<<1, 64, 0, stream>>>(sw_i, sw_j, pairSlot, nodePair, nodePartner);
  build_s0<<<4096, 256, 0, stream>>>(S_in, emb, sBufA);
  convert_x0<<<600, 256, 0, stream>>>(x_in, xbf);

  ushort_t* sCur = sBufA;
  ushort_t* sNext = sBufB;
  float* xPrev = xfA;   // unused at l=0
  float* xNext = xfA;
  for (int l = 0; l < 3; ++l) {
    int fin  = (l == 0) ? 16 : 96;
    int Kpad = (l == 0) ? 32 : 96;
    int KC   = Kpad / 32;
    int residual = (l > 0) ? 1 : 0;

    pack_proj<<<144, 256, 0, stream>>>(Wl[l][0], Wl[l][1], Wl[l][2], Wl[l][3], fin, Kpad, WprojP);
    pack_b<<<36, 256, 0, stream>>>(Wl[l][4], fin, Kpad, 96, WeP);

    // projections: (B*48 x Kpad) @ (Kpad x 384) -> proj
    gemm_rows<6><<<600, 128, 0, stream>>>(xbf, Kpad, WprojP, KC, nullptr, 0, proj, 384);
    // edge state: 460800 rows
    edge_kernel<<<28800, 64, 0, stream>>>(sCur, Kpad, KC, WeP, proj, sNext, gate, pairSlot, residual);
    // node update
    node_kernel<<<Bsz, 128, 0, stream>>>(A_in, proj, gate, nodePair, nodePartner,
                                         xPrev, residual, xNext, xbf);
    // ping-pong
    ushort_t* st = sCur; sCur = sNext; sNext = st;
    xPrev = xNext;
    xNext = (xPrev == xfA) ? xfB : xfA;
  }
  float* x3 = xPrev;        // after 3 layers
  ushort_t* s3 = sCur;

  xg_kernel<<<Bsz, 96, 0, stream>>>(x3, xg);

  // switch MLP
  build_smlp_in<<<2400, 256, 0, stream>>>(s3, x3, xg, sw_i, sw_j, smlpIn);
  pack_b<<<576, 256, 0, stream>>>(sW1, 384, 384, 384, W1sP);
  gemm_rows<6><<<100, 128, 0, stream>>>(smlpIn, 384, W1sP, 12, sb1, 1, smlpHid, 384);
  mlp2_kernel<<<50, 128, 0, stream>>>(smlpHid, 384, 4, 1600, sW2, sb2, smlpOut);

  // edge MLP
  build_cmlp_in<<<4096, 256, 0, stream>>>(x3, xg, e_i, e_j, cmlpIn);
  pack_b<<<324, 256, 0, stream>>>(cW1, 288, 288, 288, W1cP);
  gemm_rows<6><<<588, 96, 0, stream>>>(cmlpIn, 288, W1cP, 9, cb1, 1, cmlpHid, 288);
  mlp2_kernel<<<221, 128, 0, stream>>>(cmlpHid, 288, 3, Bsz * NEc, cW2, cb2, cmlpOut);

  float* z = (float*)d_out;
  float* zc = z + (size_t)Bsz * 158;
  final_kernel<<<Bsz, 64, 0, stream>>>(smlpOut, cmlpOut, Dinv, Incp, Incc, z, zc);
}
